// AnyVariateAttention_73461120631346
// MI455X (gfx1250) — matrix operand order repaired and checked by reading, not compiled, not run on hardware
//
#include <hip/hip_runtime.h>
#include <hip/hip_bf16.h>

// ---------------------------------------------------------------------------
// AnyVariateAttention for MI455X (gfx1250, wave32, WMMA bf16 16x16x32)
//   B=2, N=4096, D=256, H=8, hd=32, rope_angles=8, scaling=32^-0.5
// ---------------------------------------------------------------------------

#define SEQN   4096
#define DIM    256
#define HEADS  8
#define HD     32
#define BATCH  2
#define BH     (BATCH*HEADS)

typedef float  v8f   __attribute__((ext_vector_type(8)));
typedef __bf16 v16bf __attribute__((ext_vector_type(16)));
typedef __bf16 v8bf  __attribute__((ext_vector_type(8)));
typedef int    v4i   __attribute__((ext_vector_type(4)));

__device__ __forceinline__ v8f wmma_bf16(v16bf a, v16bf b, v8f c) {
    // D(16x16 f32) = A(16x32 bf16) * B(32x16 bf16) + C
    return __builtin_amdgcn_wmma_f32_16x16x32_bf16(
        /*neg_a=*/false, a, /*neg_b=*/false, b,
        /*c_mod=*/(short)0, c, /*reuse_a=*/false, /*reuse_b=*/false);
}

// ---- gfx1250 async global->LDS copy (ASYNCcnt-tracked), guarded ----------
#if __has_builtin(__builtin_amdgcn_global_load_async_to_lds_b128)
#define HAVE_ASYNC_LDS 1
#endif

__device__ __forceinline__ void async_copy_b128(const __bf16* g, __bf16* l) {
#ifdef HAVE_ASYNC_LDS
    __builtin_amdgcn_global_load_async_to_lds_b128(
        (__attribute__((address_space(1))) v4i*)g,
        (__attribute__((address_space(3))) v4i*)l, 0, 0);
#else
    *(v8bf*)l = *(const v8bf*)g;   // sync fallback: global->VGPR->ds_store
#endif
}

__device__ __forceinline__ void wait_async0() {
#ifdef HAVE_ASYNC_LDS
#if __has_builtin(__builtin_amdgcn_s_wait_asynccnt)
    __builtin_amdgcn_s_wait_asynccnt(0);
#else
    asm volatile("s_wait_asynccnt 0" ::: "memory");
#endif
#endif
}

// ---------------------------------------------------------------------------
// Kernel 1: fused QKV projection (+bias, +q-scaling, +partial RoPE)
//   grid.x = 8192/16 row strips; block = 256 (8 waves); wave <-> head.
//   Outputs: qh,kh as bf16 [BH, N, 32]; v transposed as bf16 [BH, 32, N].
// ---------------------------------------------------------------------------
__global__ __launch_bounds__(256)
void qkv_rope_kernel(const float* __restrict__ hidden,
                     const float* __restrict__ qw, const float* __restrict__ qb,
                     const float* __restrict__ kw, const float* __restrict__ kb,
                     const float* __restrict__ vw, const float* __restrict__ vb,
                     __bf16* __restrict__ qh, __bf16* __restrict__ kh,
                     __bf16* __restrict__ vT)
{
    const int lane = threadIdx.x & 31;
    const int wave = threadIdx.x >> 5;
    const int head = wave;
    const int half = lane >> 4;
    const int lc   = lane & 15;
    const int row0 = blockIdx.x * 16;
    const int arow = row0 + lc;

    const float* Wm[3] = { qw, kw, vw };
    v8f acc[3][2];
    #pragma unroll
    for (int m = 0; m < 3; ++m) { acc[m][0] = (v8f)0.0f; acc[m][1] = (v8f)0.0f; }

    for (int kk = 0; kk < DIM; kk += 32) {
        v16bf A;
        const float* hp = hidden + (size_t)arow * DIM + kk + half * 16;
        #pragma unroll
        for (int i = 0; i < 8; ++i) {
            A[i]     = (__bf16)hp[i];
            A[8 + i] = (__bf16)hp[8 + i];
        }
        #pragma unroll
        for (int m = 0; m < 3; ++m) {
            #pragma unroll
            for (int t = 0; t < 2; ++t) {
                const int c = head * HD + t * 16 + lc;
                const float* wp = Wm[m] + (size_t)c * DIM + kk + half * 16;
                v16bf Bf;
                #pragma unroll
                for (int i = 0; i < 16; ++i) Bf[i] = (__bf16)wp[i];
                acc[m][t] = wmma_bf16(A, Bf, acc[m][t]);
            }
        }
    }

    const int   b  = row0 / SEQN;
    const int   n0 = row0 % SEQN;
    const float scaling = 0.17677669529663687f;  // 32^-0.5

    // 1/timescale = 2^(-a*log2(1e4)/16); active angles a = lc < 8
    const float invts = __builtin_amdgcn_exp2f(-(float)lc * 0.8304820237218405f);

    const float qb0 = qb[head*HD + lc],      qb1 = qb[head*HD + 16 + lc];
    const float kb0 = kb[head*HD + lc],      kb1 = kb[head*HD + 16 + lc];
    const float vb0 = vb[head*HD + lc],      vb1 = vb[head*HD + 16 + lc];

    #pragma unroll
    for (int e = 0; e < 8; ++e) {
        const int n   = n0 + e + half * 8;
        const float pos = (float)n;
        float s = 0.0f, c = 1.0f;
        if (lc < 8) { const float ang = pos * invts; __sincosf(ang, &s, &c); }

        float q0 = (acc[0][0][e] + qb0) * scaling;
        float q1 = (acc[0][1][e] + qb1) * scaling;
        if (lc < 8) { const float f = q0, sd = q1; q0 = f*c - sd*s; q1 = sd*c + f*s; }
        const size_t qkbase = ((size_t)(b*HEADS + head) * SEQN + n) * HD;
        qh[qkbase + lc]      = (__bf16)q0;
        qh[qkbase + 16 + lc] = (__bf16)q1;

        float k0 = acc[1][0][e] + kb0;
        float k1 = acc[1][1][e] + kb1;
        if (lc < 8) { const float f = k0, sd = k1; k0 = f*c - sd*s; k1 = sd*c + f*s; }
        kh[qkbase + lc]      = (__bf16)k0;
        kh[qkbase + 16 + lc] = (__bf16)k1;

        const size_t vb_base = (size_t)(b*HEADS + head) * HD * SEQN;
        vT[vb_base + (size_t)(lc)      * SEQN + n] = (__bf16)(acc[2][0][e] + vb0);
        vT[vb_base + (size_t)(16 + lc) * SEQN + n] = (__bf16)(acc[2][1][e] + vb1);
    }
}

// ---------------------------------------------------------------------------
// Kernel 2: fused flash attention with variate bias.
//   grid = (N/128, BH); block = 256 (8 waves); wave owns 16 q rows.
//   kv streamed in chunks of 64, double-buffered in LDS via async DMA:
//     4 WMMA (S) + online softmax + 4 WMMA (PV) per chunk.
// ---------------------------------------------------------------------------
#define KVCH 64

__global__ __launch_bounds__(256)
void flash_attn_kernel(const __bf16* __restrict__ qh,
                       const __bf16* __restrict__ kh,
                       const __bf16* __restrict__ vT,
                       const float*  __restrict__ attn_bias, // [H,2]
                       const int*    __restrict__ seq_ptr,
                       float* __restrict__ ctx)              // [B,N,D] f32
{
    __shared__ __bf16 KtBuf[2][KVCH * HD];   // 2 x 4KB : K rows  [64][32]
    __shared__ __bf16 VtBuf[2][HD * KVCH];   // 2 x 4KB : V^T     [32][64]
    __shared__ __bf16 Plds[8][16 * KVCH];    // 16KB    : per-wave P staging

    const int tid  = threadIdx.x;
    const int lane = tid & 31;
    const int wave = tid >> 5;
    const int half = lane >> 4;
    const int lc   = lane & 15;
    const int bh   = blockIdx.y;
    const int b    = bh >> 3, h = bh & 7;
    const int row0 = blockIdx.x * 128 + wave * 16;

    const __bf16* Q = qh + (size_t)bh * SEQN * HD;
    const __bf16* K = kh + (size_t)bh * SEQN * HD;
    const __bf16* V = vT + (size_t)bh * HD * SEQN;

    const int   seq   = seq_ptr[0];
    const float bsame = attn_bias[h * 2 + 0];
    const float bdiff = attn_bias[h * 2 + 1];

    // Q A-fragment, held for the whole kv loop
    v16bf aQ;
    {
        const __bf16* p = Q + (size_t)(row0 + lc) * HD + half * 16;
        #pragma unroll
        for (int i = 0; i < 8; ++i) { aQ[i] = p[i]; aQ[8 + i] = p[8 + i]; }
    }

    v8f o0 = (v8f)0.0f, o1 = (v8f)0.0f;
    float m[8], l[8];
    int   qv[8];
    #pragma unroll
    for (int e = 0; e < 8; ++e) {
        m[e] = -1e30f; l[e] = 0.0f;
        qv[e] = (row0 + e + half * 8) / seq;
    }

    __bf16* Pw = Plds[wave];
    const int ktop = half * 16;

    // per-thread DMA assignments (one b128 each for K tile and V tile)
    const int krow = tid >> 2, kseg = (tid & 3) * 8;   // K: [64 rows][32]
    const int vrow = tid >> 3, vseg = (tid & 7) * 8;   // V^T: [32 rows][64]

    // prologue: stage chunk 0 into buffer 0
    async_copy_b128(K + (size_t)(0 + krow) * HD + kseg, &KtBuf[0][krow * HD + kseg]);
    async_copy_b128(V + (size_t)vrow * SEQN + 0 + vseg, &VtBuf[0][vrow * KVCH + vseg]);

    for (int c0 = 0; c0 < SEQN; c0 += KVCH) {
        const int cur = (c0 / KVCH) & 1;

        wait_async0();        // my DMA for buffer `cur` has landed in LDS
        __syncthreads();      // everyone's DMA landed; prev compute done

        // prefetch next chunk into the other buffer (disjoint from `cur`)
        const int c1 = c0 + KVCH;
        if (c1 < SEQN) {
            async_copy_b128(K + (size_t)(c1 + krow) * HD + kseg,
                            &KtBuf[1 - cur][krow * HD + kseg]);
            async_copy_b128(V + (size_t)vrow * SEQN + c1 + vseg,
                            &VtBuf[1 - cur][vrow * KVCH + vseg]);
        }

        const __bf16* Kt = KtBuf[cur];
        const __bf16* Vt = VtBuf[cur];

        // ---- S = Q K^T : four 16x16 tiles over kv cols [c0, c0+64) ----
        v8f s4[4];
        #pragma unroll
        for (int j = 0; j < 4; ++j) {
            const v16bf bK = *(const v16bf*)(Kt + (j * 16 + lc) * HD + ktop);
            s4[j] = wmma_bf16(aQ, bK, (v8f)0.0f);
        }

        int kvv[4];
        #pragma unroll
        for (int j = 0; j < 4; ++j) kvv[j] = (c0 + j * 16 + lc) / seq;

        // ---- online softmax (row stats across 16 lanes of each half) ----
        #pragma unroll
        for (int e = 0; e < 8; ++e) {
            float a[4];
            #pragma unroll
            for (int j = 0; j < 4; ++j)
                a[j] = s4[j][e] + ((qv[e] == kvv[j]) ? bsame : bdiff);

            float t = fmaxf(fmaxf(a[0], a[1]), fmaxf(a[2], a[3]));
            t = fmaxf(t, __shfl_xor(t, 1, 32));
            t = fmaxf(t, __shfl_xor(t, 2, 32));
            t = fmaxf(t, __shfl_xor(t, 4, 32));
            t = fmaxf(t, __shfl_xor(t, 8, 32));
            const float mn = fmaxf(m[e], t);
            const float sc = __expf(m[e] - mn);

            float p[4], rs = 0.0f;
            #pragma unroll
            for (int j = 0; j < 4; ++j) { p[j] = __expf(a[j] - mn); rs += p[j]; }
            rs += __shfl_xor(rs, 1, 32);
            rs += __shfl_xor(rs, 2, 32);
            rs += __shfl_xor(rs, 4, 32);
            rs += __shfl_xor(rs, 8, 32);

            l[e] = l[e] * sc + rs;
            m[e] = mn;
            o0[e] *= sc; o1[e] *= sc;

            const int r = e + half * 8;
            #pragma unroll
            for (int j = 0; j < 4; ++j)
                Pw[r * KVCH + j * 16 + lc] = (__bf16)p[j];
        }

        // ---- reload P as two A fragments (16x64, intra-wave LDS) ----
        v16bf aP0, aP1;
        {
            const __bf16* p = Pw + lc * KVCH + half * 16;
            #pragma unroll
            for (int i = 0; i < 8; ++i) {
                aP0[i] = p[i];      aP0[8 + i] = p[8 + i];
                aP1[i] = p[32 + i]; aP1[8 + i] = p[40 + i];
            }
        }

        // ---- O += P V : 2 d-tiles x 2 k-steps ----
        #pragma unroll
        for (int s = 0; s < 2; ++s) {
            const v16bf aP = s ? aP1 : aP0;
            const v16bf bV0 = *(const v16bf*)(Vt + (lc)      * KVCH + s * 32 + ktop);
            const v16bf bV1 = *(const v16bf*)(Vt + (16 + lc) * KVCH + s * 32 + ktop);
            o0 = wmma_bf16(aP, bV0, o0);
            o1 = wmma_bf16(aP, bV1, o1);
        }
    }

    // normalize and write context in [B, N, H*hd] layout (ready for O-proj)
    #pragma unroll
    for (int e = 0; e < 8; ++e) {
        const int n = row0 + e + half * 8;
        const float inv = 1.0f / l[e];
        const size_t base = ((size_t)b * SEQN + n) * DIM + h * HD;
        ctx[base + lc]      = o0[e] * inv;
        ctx[base + 16 + lc] = o1[e] * inv;
    }
}

// ---------------------------------------------------------------------------
// Kernel 3: output projection  out = ctx @ o_w^T + o_b
//   grid = (8192/16, 2); block = 256; wave owns a 16x16 output tile.
// ---------------------------------------------------------------------------
__global__ __launch_bounds__(256)
void oproj_kernel(const float* __restrict__ ctx,
                  const float* __restrict__ ow, const float* __restrict__ ob,
                  float* __restrict__ out)
{
    const int lane = threadIdx.x & 31;
    const int wave = threadIdx.x >> 5;
    const int half = lane >> 4;
    const int lc   = lane & 15;
    const int row0 = blockIdx.x * 16;
    const int c0   = blockIdx.y * 128 + wave * 16;
    const int arow = row0 + lc;

    v8f acc = (v8f)0.0f;
    for (int kk = 0; kk < DIM; kk += 32) {
        v16bf A;
        const float* hp = ctx + (size_t)arow * DIM + kk + half * 16;
        #pragma unroll
        for (int i = 0; i < 8; ++i) { A[i] = (__bf16)hp[i]; A[8+i] = (__bf16)hp[8+i]; }
        v16bf Bf;
        const float* wp = ow + (size_t)(c0 + lc) * DIM + kk + half * 16;
        #pragma unroll
        for (int i = 0; i < 16; ++i) Bf[i] = (__bf16)wp[i];
        acc = wmma_bf16(A, Bf, acc);
    }

    const float bias = ob[c0 + lc];
    #pragma unroll
    for (int e = 0; e < 8; ++e) {
        out[(size_t)(row0 + e + half * 8) * DIM + c0 + lc] = acc[e] + bias;
    }
}

// ---------------------------------------------------------------------------
extern "C" void kernel_launch(void* const* d_in, const int* in_sizes, int n_in,
                              void* d_out, int out_size, void* d_ws, size_t ws_size,
                              hipStream_t stream)
{
    const float* hidden = (const float*)d_in[0];
    const float* qw = (const float*)d_in[1];
    const float* qb = (const float*)d_in[2];
    const float* kw = (const float*)d_in[3];
    const float* kb = (const float*)d_in[4];
    const float* vw = (const float*)d_in[5];
    const float* vb = (const float*)d_in[6];
    const float* ow = (const float*)d_in[7];
    const float* ob = (const float*)d_in[8];
    const float* ab = (const float*)d_in[9];
    const int* seqp = (const int*)d_in[11];

    char* ws = (char*)d_ws;
    const size_t MB = 1024 * 1024;
    __bf16* qh  = (__bf16*)(ws + 0 * MB);   // 4 MB  [BH,N,32] bf16
    __bf16* kh  = (__bf16*)(ws + 4 * MB);   // 4 MB  [BH,N,32] bf16
    __bf16* vT  = (__bf16*)(ws + 8 * MB);   // 4 MB  [BH,32,N] bf16
    float*  ctx = (float*)(ws + 12 * MB);   // 8 MB  [B,N,D]   f32

    qkv_rope_kernel<<<dim3(BATCH * SEQN / 16), 256, 0, stream>>>(
        hidden, qw, qb, kw, kb, vw, vb, qh, kh, vT);

    flash_attn_kernel<<<dim3(SEQN / 128, BH), 256, 0, stream>>>(
        qh, kh, vT, ab, seqp, ctx);

    oproj_kernel<<<dim3(BATCH * SEQN / 16, 2), 256, 0, stream>>>(
        ctx, ow, ob, (float*)d_out);
}
